// GraphAttentionLayer_6777458393270
// MI455X (gfx1250) — matrix operand order repaired and checked by reading, not compiled, not run on hardware
//
#include <hip/hip_runtime.h>

// ---------------------------------------------------------------------------
// GAT layer for MI455X (gfx1250).
//   feat[N,256] @ W[256,128] via v_wmma_f32_16x16x32_bf16 with bf16 hi/lo
//   split (3 WMMAs ~ fp32 accuracy), then L2-resident edge softmax/aggregate.
// ---------------------------------------------------------------------------

#define N_NODES 100000
#define N_EDGES 1600000
#define IN_DIM  256
#define OUT_DIM 32
#define N_HEADS 4
#define HD      128          // N_HEADS * OUT_DIM
#define NEG_SLOPE 0.2f
#define N_TILES (N_NODES / 16)   // 6250, exact

typedef __bf16 v16bf __attribute__((ext_vector_type(16)));
typedef float  v8f   __attribute__((ext_vector_type(8)));

// ---- float atomic max via sign-split integer atomics (no CAS loop) --------
__device__ __forceinline__ void atomicMaxF(float* addr, float v) {
    if (v >= 0.0f) atomicMax((int*)addr, __float_as_int(v));
    else           atomicMin((unsigned int*)addr, __float_as_uint(v));
}

// ---- init: rst = 0, e_max = -inf, e_sum = 0 -------------------------------
__global__ void gat_init(float* __restrict__ rst, float* __restrict__ emax,
                         float* __restrict__ esum) {
    int idx = blockIdx.x * blockDim.x + threadIdx.x;
    if (idx < N_NODES * HD) rst[idx] = 0.0f;
    if (idx < N_NODES * N_HEADS) {
        emax[idx] = __int_as_float(0xFF800000);  // -inf
        esum[idx] = 0.0f;
    }
}

// ---- pack W[256,128] into per-lane WMMA B-operand layout, bf16 hi/lo ------
// B operand as packed here (16-bit, 32x16): lane&15 = column n, lane>>4 selects K half,
// element j of v16bf = K = (lane>>4)*16 + j (contiguous order chosen here, not AMD's table order; A matches).
// Packed index: [kc(0..7)][coltile t(0..7)][lane(0..31)][j(0..15)]
__global__ void gat_pack_w(const float* __restrict__ W,
                           __bf16* __restrict__ whi, __bf16* __restrict__ wlo) {
    int id = blockIdx.x * blockDim.x + threadIdx.x;
    if (id >= 8 * 8 * 32) return;
    int lane = id & 31;
    int t    = (id >> 5) & 7;
    int kc   = id >> 8;
    int kbase = kc * 32 + (lane >> 4) * 16;
    int col   = t * 16 + (lane & 15);
#pragma unroll
    for (int j = 0; j < 16; ++j) {
        float v = W[(kbase + j) * HD + col];
        __bf16 h = (__bf16)v;
        whi[id * 16 + j] = h;
        wlo[id * 16 + j] = (__bf16)(v - (float)h);
    }
}

// ---- projection GEMM: ft[N,128] = feat[N,256] @ W, split-bf16 WMMA --------
// One wave per 16-row tile; 8 column tiles of 16; K in 8 chunks of 32.
__global__ __launch_bounds__(256) void gat_gemm_wmma(
        const float* __restrict__ feat, const __bf16* __restrict__ whi,
        const __bf16* __restrict__ wlo, float* __restrict__ ft) {
    const int lane = threadIdx.x & 31;
    const int wave = threadIdx.x >> 5;
    const int tile = blockIdx.x * 8 + wave;
    if (tile >= N_TILES) return;                 // wave-uniform: EXEC stays full
    const int row0 = tile * 16;
    const int m    = lane & 15;                  // A row
    const int kb   = (lane >> 4) * 16;           // A K sub-offset
    const float* arow = feat + (size_t)(row0 + m) * IN_DIM;
    const v16bf* bh = (const v16bf*)whi;
    const v16bf* bl = (const v16bf*)wlo;

    v8f acc[8] = {};

#pragma unroll
    for (int kc = 0; kc < 8; ++kc) {
        const int k0 = kc * 32;
        float va[16];
#pragma unroll
        for (int j = 0; j < 8; ++j) va[j]     = arow[k0 + kb + j];
#pragma unroll
        for (int j = 0; j < 8; ++j) va[8 + j] = arow[k0 + kb + 8 + j];
        v16bf ahi, alo;
#pragma unroll
        for (int j = 0; j < 16; ++j) {
            __bf16 h = (__bf16)va[j];
            ahi[j] = h;
            alo[j] = (__bf16)(va[j] - (float)h);
        }
#pragma unroll
        for (int t = 0; t < 8; ++t) {
            v16bf bhv = bh[(kc * 8 + t) * 32 + lane];
            v16bf blv = bl[(kc * 8 + t) * 32 + lane];
            // A*B ~= Alo*Bhi + Ahi*Blo + Ahi*Bhi  (fp32 accumulate)
            acc[t] = __builtin_amdgcn_wmma_f32_16x16x32_bf16(
                         false, alo, false, bhv, (short)0, acc[t], false, false);
            acc[t] = __builtin_amdgcn_wmma_f32_16x16x32_bf16(
                         false, ahi, false, blv, (short)0, acc[t], false, false);
            acc[t] = __builtin_amdgcn_wmma_f32_16x16x32_bf16(
                         false, ahi, false, bhv, (short)0, acc[t], false, false);
        }
    }

    // C/D layout: VGPR j -> M = j + 8*(lane>>4), N = lane&15
    const int mrow = 8 * (lane >> 4);
    const int ncol = lane & 15;
#pragma unroll
    for (int t = 0; t < 8; ++t)
#pragma unroll
        for (int j = 0; j < 8; ++j)
            ft[(size_t)(row0 + mrow + j) * HD + t * 16 + ncol] = acc[t][j];
}

// ---- per-node attention logits el, er [N, H] ------------------------------
__global__ void gat_elr(const float* __restrict__ ft,
                        const float* __restrict__ attn_l,
                        const float* __restrict__ attn_r,
                        float* __restrict__ el, float* __restrict__ er) {
    int idx = blockIdx.x * blockDim.x + threadIdx.x;
    if (idx >= N_NODES * N_HEADS) return;
    int n = idx >> 2, h = idx & 3;
    const float* f  = ft + (size_t)n * HD + h * OUT_DIM;
    const float* al = attn_l + h * OUT_DIM;
    const float* ar = attn_r + h * OUT_DIM;
    float sl = 0.0f, sr = 0.0f;
#pragma unroll
    for (int d = 0; d < OUT_DIM; ++d) {
        float x = f[d];
        sl += x * al[d];
        sr += x * ar[d];
    }
    el[idx] = sl;
    er[idx] = sr;
}

// ---- edge pass A: e = leaky_relu(el[src]+er[dst]); segment max ------------
__global__ void gat_edge_max(const int* __restrict__ src, const int* __restrict__ dst,
                             const float* __restrict__ el, const float* __restrict__ er,
                             float* __restrict__ eedge, float* __restrict__ emax) {
    int idx = blockIdx.x * blockDim.x + threadIdx.x;
    if (idx >= N_EDGES * N_HEADS) return;
    int e = idx >> 2, h = idx & 3;
    int s = src[e], d = dst[e];
    float v = el[s * N_HEADS + h] + er[d * N_HEADS + h];
    v = (v > 0.0f) ? v : NEG_SLOPE * v;
    eedge[idx] = v;
    atomicMaxF(&emax[d * N_HEADS + h], v);
}

// ---- edge pass B: e_exp = exp(e - max[dst]); segment sum ------------------
__global__ void gat_edge_exp(const int* __restrict__ dst,
                             const float* __restrict__ emax,
                             float* __restrict__ eedge, float* __restrict__ esum) {
    int idx = blockIdx.x * blockDim.x + threadIdx.x;
    if (idx >= N_EDGES * N_HEADS) return;
    int e = idx >> 2, h = idx & 3;
    int d = dst[e];
    float ex = __expf(eedge[idx] - emax[d * N_HEADS + h]);
    eedge[idx] = ex;                    // in-place: now holds exp value
    atomicAdd(&esum[d * N_HEADS + h], ex);
}

// ---- edge pass C: rst[dst] += ft[src] * (e_exp / e_sum[dst]) --------------
// 128 threads per edge, 2 edges per block; ft & rst are L2-resident (51MB ea).
__global__ __launch_bounds__(256) void gat_edge_aggr(
        const int* __restrict__ src, const int* __restrict__ dst,
        const float* __restrict__ eedge, const float* __restrict__ esum,
        const float* __restrict__ ft, float* __restrict__ rst) {
    int e = blockIdx.x * 2 + (threadIdx.x >> 7);
    int i = threadIdx.x & 127;
    if (e >= N_EDGES) return;
    int s = src[e], d = dst[e];
    int h = i >> 5;
    float a = eedge[e * N_HEADS + h] / esum[d * N_HEADS + h];
    atomicAdd(&rst[(size_t)d * HD + i], ft[(size_t)s * HD + i] * a);
}

// ---- finalize: out[n,d] = mean over heads ---------------------------------
__global__ void gat_mean(const float* __restrict__ rst, float* __restrict__ out) {
    int idx = blockIdx.x * blockDim.x + threadIdx.x;
    if (idx >= N_NODES * OUT_DIM) return;
    int n = idx >> 5, d = idx & 31;
    const float* r = rst + (size_t)n * HD;
    out[idx] = 0.25f * (r[d] + r[32 + d] + r[64 + d] + r[96 + d]);
}

// ---------------------------------------------------------------------------
extern "C" void kernel_launch(void* const* d_in, const int* in_sizes, int n_in,
                              void* d_out, int out_size, void* d_ws, size_t ws_size,
                              hipStream_t stream) {
    (void)in_sizes; (void)n_in; (void)out_size; (void)ws_size;
    const float* feat   = (const float*)d_in[0];
    const float* W      = (const float*)d_in[1];
    const float* attn_l = (const float*)d_in[2];
    const float* attn_r = (const float*)d_in[3];
    const int*   src    = (const int*)d_in[4];
    const int*   dst    = (const int*)d_in[5];
    float* out = (float*)d_out;

    // workspace carve-up (~134.6 MB total)
    char* p = (char*)d_ws;
    float* ft    = (float*)p;  p += (size_t)N_NODES * HD * sizeof(float);
    float* rst   = (float*)p;  p += (size_t)N_NODES * HD * sizeof(float);
    float* el    = (float*)p;  p += (size_t)N_NODES * N_HEADS * sizeof(float);
    float* er    = (float*)p;  p += (size_t)N_NODES * N_HEADS * sizeof(float);
    float* emax  = (float*)p;  p += (size_t)N_NODES * N_HEADS * sizeof(float);
    float* esum  = (float*)p;  p += (size_t)N_NODES * N_HEADS * sizeof(float);
    float* eedge = (float*)p;  p += (size_t)N_EDGES * N_HEADS * sizeof(float);
    __bf16* whi  = (__bf16*)p; p += (size_t)2048 * 16 * sizeof(__bf16);
    __bf16* wlo  = (__bf16*)p;

    gat_init<<<(N_NODES * HD + 255) / 256, 256, 0, stream>>>(rst, emax, esum);
    gat_pack_w<<<8, 256, 0, stream>>>(W, whi, wlo);
    gat_gemm_wmma<<<(N_TILES + 7) / 8, 256, 0, stream>>>(feat, whi, wlo, ft);
    gat_elr<<<(N_NODES * N_HEADS + 255) / 256, 256, 0, stream>>>(ft, attn_l, attn_r, el, er);
    gat_edge_max<<<(N_EDGES * N_HEADS + 255) / 256, 256, 0, stream>>>(src, dst, el, er, eedge, emax);
    gat_edge_exp<<<(N_EDGES * N_HEADS + 255) / 256, 256, 0, stream>>>(dst, emax, eedge, esum);
    gat_edge_aggr<<<N_EDGES / 2, 256, 0, stream>>>(src, dst, eedge, esum, ft, rst);
    gat_mean<<<(N_NODES * OUT_DIM + 255) / 256, 256, 0, stream>>>(rst, out);
}
